// LGCN2_83528523973374
// MI455X (gfx1250) — matrix operand order repaired and checked by reading, not compiled, not run on hardware
//
#include <hip/hip_runtime.h>
#include <hip/hip_bf16.h>

typedef __attribute__((ext_vector_type(16))) _Float16 v16h;
typedef __attribute__((ext_vector_type(4)))  _Float16 v4h;
typedef __attribute__((ext_vector_type(8)))  float    v8f;
typedef __attribute__((ext_vector_type(2)))  float    v2f;
typedef __attribute__((ext_vector_type(4)))  float    v4f;

#define NT_   8000
#define R_    8000
#define N_    10000
#define RP_   16
#define E_    64
#define C_    16
#define LW_   64
#define NKB_  250      // R_/32 k-blocks for f16 WMMA

// ---------------------------------------------------------------------------
// K0: build fragment-swizzled f16 B = [w1a | w2a]  (R_ x 128)
// Fragment layout (32x16 f16 B): lane l: n = l%16 ; half h: K = (h<8?0:16)+(l/16)*8+(h&7)
// Stored so a wave's lane reads one contiguous v16h per k-block.
// ---------------------------------------------------------------------------
__global__ __launch_bounds__(256) void k_buildB(const float* __restrict__ w1a,
                                                const float* __restrict__ w2a,
                                                _Float16* __restrict__ Bsw)
{
    int idx  = blockIdx.x * 256 + threadIdx.x;      // 0 .. 8000*128-1
    int h    = idx & 15;
    int lane = (idx >> 4) & 31;
    int rest = idx >> 9;                            // j*NKB_ + kb
    int kb   = rest % NKB_;
    int j    = rest / NKB_;
    int K    = kb * 32 + ((h >> 3) << 4) + ((lane >> 4) << 3) + (h & 7);
    int n    = j * 16 + (lane & 15);
    float v  = (n < LW_) ? w1a[K * LW_ + n] : w2a[K * LW_ + (n - LW_)];
    Bsw[idx] = (_Float16)v;
}

// ---------------------------------------------------------------------------
// K1: P = relu(nhots @ [w1a|w2a] + [b1a|b2a])  via v_wmma_f32_16x16x32_f16
// 250 workgroups (32-row strips), 8 waves, each wave owns one 16-col tile and
// two 16-row blocks (acc0/acc1).  Ping-pong LDS, one barrier per 64-K step.
// A fragment layout: lane l: M = l%16 ; half h: K = (h<8?0:16)+(l/16)*8+(h&7)
// LDS layout: [(kstep*2+rb)*32 + lane][h], groups of 4 K -> contiguous halves
// ---------------------------------------------------------------------------
__global__ __launch_bounds__(256) void k_stageA(const float* __restrict__ nhots,
                                                const _Float16* __restrict__ Bsw,
                                                const float* __restrict__ b1a,
                                                const float* __restrict__ b2a,
                                                float* __restrict__ P1,
                                                float* __restrict__ P2)
{
    __shared__ _Float16 ldsA[2][2048];              // ping-pong, 4 KB each
    const int tid  = threadIdx.x;
    const int wave = tid >> 5;
    const int lane = tid & 31;
    const v16h* bfr = (const v16h*)Bsw;

    // per-thread fill coordinates (one float4 per row-block per k-step)
    const int e4    = tid * 4;                      // 0..1020
    const int row16 = (e4 >> 6) & 15;
    const int kk    = e4 & 63;                      // multiple of 4
    const int kstep = kk >> 5;
    const int kk32  = kk & 31;
    const int hif   = (kk32 >> 3) & 1;
    const int h0    = ((kk32 >> 4) << 3) | (kk32 & 7);   // multiple of 4
    const int ldst  = ((kstep * 2) * 32 + (hif * 16 + row16)) * 16 + h0;
    const float* g0 = nhots + (size_t)(blockIdx.x * 32 + row16) * R_ + kk;
    const float* g1 = g0 + (size_t)16 * R_;

    v8f acc0 = {}, acc1 = {};

    {   // preload k-step 0
        v4f x0 = *(const v4f*)g0;
        v4f x1 = *(const v4f*)g1;
        v4h p0, p1;
        p0.x = (_Float16)x0.x; p0.y = (_Float16)x0.y;
        p0.z = (_Float16)x0.z; p0.w = (_Float16)x0.w;
        p1.x = (_Float16)x1.x; p1.y = (_Float16)x1.y;
        p1.z = (_Float16)x1.z; p1.w = (_Float16)x1.w;
        *(v4h*)&ldsA[0][ldst]       = p0;
        *(v4h*)&ldsA[0][ldst + 512] = p1;
    }
    __syncthreads();

    for (int it = 0; it < 125; ++it) {
        const int buf = it & 1;
        v4f x0, x1;
        if (it < 124) {                              // prefetch next strip
            x0 = *(const v4f*)(g0 + (it + 1) * 64);
            x1 = *(const v4f*)(g1 + (it + 1) * 64);
        }
        const _Float16* base = &ldsA[buf][0];
        v16h a00 = *(const v16h*)&base[(0 * 32 + lane) * 16];   // ks0 rb0
        v16h a01 = *(const v16h*)&base[(1 * 32 + lane) * 16];   // ks0 rb1
        v16h a10 = *(const v16h*)&base[(2 * 32 + lane) * 16];   // ks1 rb0
        v16h a11 = *(const v16h*)&base[(3 * 32 + lane) * 16];   // ks1 rb1
        const int kb = it * 2;
        v16h b0 = bfr[(wave * NKB_ + kb)     * 32 + lane];
        v16h b1 = bfr[(wave * NKB_ + kb + 1) * 32 + lane];
        acc0 = __builtin_amdgcn_wmma_f32_16x16x32_f16(false, a00, false, b0,
                                                      (short)0, acc0, false, false);
        acc1 = __builtin_amdgcn_wmma_f32_16x16x32_f16(false, a01, false, b0,
                                                      (short)0, acc1, false, false);
        acc0 = __builtin_amdgcn_wmma_f32_16x16x32_f16(false, a10, false, b1,
                                                      (short)0, acc0, false, false);
        acc1 = __builtin_amdgcn_wmma_f32_16x16x32_f16(false, a11, false, b1,
                                                      (short)0, acc1, false, false);
        if (it < 124) {                              // fill other buffer
            v4h p0, p1;
            p0.x = (_Float16)x0.x; p0.y = (_Float16)x0.y;
            p0.z = (_Float16)x0.z; p0.w = (_Float16)x0.w;
            p1.x = (_Float16)x1.x; p1.y = (_Float16)x1.y;
            p1.z = (_Float16)x1.z; p1.w = (_Float16)x1.w;
            *(v4h*)&ldsA[buf ^ 1][ldst]       = p0;
            *(v4h*)&ldsA[buf ^ 1][ldst + 512] = p1;
        }
        __syncthreads();
    }

    const int   jcol = wave * 16 + (lane & 15);
    const int   hi   = lane >> 4;
    const float bias = (jcol < LW_) ? b1a[jcol] : b2a[jcol - LW_];
    float*      dst  = (jcol < LW_) ? P1 : P2;
    const int   jj   = (jcol < LW_) ? jcol : (jcol - LW_);
#pragma unroll
    for (int v = 0; v < 8; ++v) {
        int t0 = blockIdx.x * 32 + hi * 8 + v;       // D: M = v + (l/16)*8
        dst[t0 * LW_ + jj]        = fmaxf(acc0[v] + bias, 0.0f);
        dst[(t0 + 16) * LW_ + jj] = fmaxf(acc1[v] + bias, 0.0f);
    }
}

// ---------------------------------------------------------------------------
// K2: lat = softmax(P @ wb + bb) over the 16-wide RP axis, one thread per row
// stored as lat[t][r]
// ---------------------------------------------------------------------------
__global__ __launch_bounds__(256) void k_latents(const float* __restrict__ P1,
                                                 const float* __restrict__ P2,
                                                 const float* __restrict__ w1b,
                                                 const float* __restrict__ b1b,
                                                 const float* __restrict__ w2b,
                                                 const float* __restrict__ b2b,
                                                 float* __restrict__ lat1,
                                                 float* __restrict__ lat2)
{
    __shared__ float w1s[LW_ * 16], w2s[LW_ * 16], bb1[16], bb2[16];
    int tid = threadIdx.x;
    for (int i = tid; i < LW_ * 16; i += 256) { w1s[i] = w1b[i]; w2s[i] = w2b[i]; }
    if (tid < 16) { bb1[tid] = b1b[tid]; bb2[tid] = b2b[tid]; }
    __syncthreads();
    int t = blockIdx.x * 256 + tid;
    if (t >= NT_) return;

    float lg[16];
#pragma unroll
    for (int c = 0; c < 16; ++c) lg[c] = bb1[c];
    for (int k = 0; k < LW_; ++k) {
        float p = P1[t * LW_ + k];
#pragma unroll
        for (int c = 0; c < 16; ++c) lg[c] = fmaf(p, w1s[k * 16 + c], lg[c]);
    }
    float mx = lg[0];
#pragma unroll
    for (int c = 1; c < 16; ++c) mx = fmaxf(mx, lg[c]);
    float sum = 0.f;
#pragma unroll
    for (int c = 0; c < 16; ++c) { lg[c] = __expf(lg[c] - mx); sum += lg[c]; }
    float inv = 1.0f / sum;
#pragma unroll
    for (int c = 0; c < 16; ++c) lat1[t * 16 + c] = lg[c] * inv;

#pragma unroll
    for (int c = 0; c < 16; ++c) lg[c] = bb2[c];
    for (int k = 0; k < LW_; ++k) {
        float p = P2[t * LW_ + k];
#pragma unroll
        for (int c = 0; c < 16; ++c) lg[c] = fmaf(p, w2s[k * 16 + c], lg[c]);
    }
    mx = lg[0];
#pragma unroll
    for (int c = 1; c < 16; ++c) mx = fmaxf(mx, lg[c]);
    sum = 0.f;
#pragma unroll
    for (int c = 0; c < 16; ++c) { lg[c] = __expf(lg[c] - mx); sum += lg[c]; }
    inv = 1.0f / sum;
#pragma unroll
    for (int c = 0; c < 16; ++c) lat2[t * 16 + c] = lg[c] * inv;
}

// ---------------------------------------------------------------------------
// K3: zero contiguous scratch (colsum|rowsum|h|h2)
// ---------------------------------------------------------------------------
__global__ void k_zero(float* __restrict__ p, long n)
{
    long i4 = ((long)blockIdx.x * 256 + threadIdx.x) * 4;
    if (i4 + 3 < n) *(v4f*)(p + i4) = (v4f){0.f, 0.f, 0.f, 0.f};
}

// ---------------------------------------------------------------------------
// K4: colsum[o*r] += lat1[t][r], rowsum[s*r] += lat2[t][r]
// r==0 (hot index 0) handled with a block reduction -> one atomic per block
// ---------------------------------------------------------------------------
__global__ __launch_bounds__(256) void k_normsum(const float* __restrict__ lat1,
                                                 const float* __restrict__ lat2,
                                                 const int* __restrict__ hind,
                                                 const int* __restrict__ vind,
                                                 float* __restrict__ colsum,
                                                 float* __restrict__ rowsum)
{
    __shared__ float redc[256], redr[256];
    int tid = threadIdx.x;
    int t   = blockIdx.x * 256 + tid;
    float c0 = 0.f, r0 = 0.f;
    if (t < NT_) {
        int s = hind[2 * t];        // hrow at r=0 -> s_t
        int o = vind[2 * t + 1];    // vcol at r=0 -> o_t
        c0 = lat1[t * 16];
        r0 = lat2[t * 16];
#pragma unroll
        for (int r = 1; r < 16; ++r) {
            unsafeAtomicAdd(&colsum[o * r], lat1[t * 16 + r]);
            unsafeAtomicAdd(&rowsum[s * r], lat2[t * 16 + r]);
        }
    }
    redc[tid] = c0; redr[tid] = r0;
    __syncthreads();
    for (int off = 128; off > 0; off >>= 1) {
        if (tid < off) { redc[tid] += redc[tid + off]; redr[tid] += redr[tid + off]; }
        __syncthreads();
    }
    if (tid == 0) {
        unsafeAtomicAdd(&colsum[0], redc[0]);
        unsafeAtomicAdd(&rowsum[0], redr[0]);
    }
}

// ---------------------------------------------------------------------------
// K5: h[s_t] += sum_r (lat1[t][r]/colsum[o*r]) * weights1_flat[o*r]
// one wave per t, lane covers 2 of 64 embedding cols
// ---------------------------------------------------------------------------
__global__ __launch_bounds__(256) void k_layer1(const float* __restrict__ lat1,
                                                const float* __restrict__ colsum,
                                                const int* __restrict__ hind,
                                                const int* __restrict__ vind,
                                                const float* __restrict__ W1,
                                                float* __restrict__ h)
{
    int wave = threadIdx.x >> 5, lane = threadIdx.x & 31;
    int t = blockIdx.x * 8 + wave;
    int s = hind[2 * t];
    int o = vind[2 * t + 1];
    float a0 = 0.f, a1 = 0.f;
#pragma unroll
    for (int r = 0; r < 16; ++r) {
        int m = o * r;
        float lw = lat1[t * 16 + r] / colsum[m];
        const float* xr = W1 + (size_t)m * 64;
        a0 = fmaf(lw, xr[lane], a0);
        a1 = fmaf(lw, xr[lane + 32], a1);
    }
    unsafeAtomicAdd(&h[s * 64 + lane], a0);
    unsafeAtomicAdd(&h[s * 64 + lane + 32], a1);
}

// K6: h = relu(h + bias1)
__global__ void k_relu_bias(float* __restrict__ h, const float* __restrict__ bias1)
{
    int i = blockIdx.x * 256 + threadIdx.x;
    if (i < N_ * 64) h[i] = fmaxf(h[i] + bias1[i & 63], 0.f);
}

// ---------------------------------------------------------------------------
// K7: h2[s*r] += (lat2[t][r]/rowsum[s*r]) * h[o]
// r==0 scatters all hit h2[0]: pre-reduce per block in LDS
// ---------------------------------------------------------------------------
__global__ __launch_bounds__(256) void k_layer2(const float* __restrict__ lat2,
                                                const float* __restrict__ rowsum,
                                                const int* __restrict__ hind,
                                                const int* __restrict__ vind,
                                                const float* __restrict__ h,
                                                float* __restrict__ h2)
{
    __shared__ float acc0[64];
    int tid = threadIdx.x;
    if (tid < 64) acc0[tid] = 0.f;
    __syncthreads();
    int wave = tid >> 5, lane = tid & 31;
    int t = blockIdx.x * 8 + wave;
    int s = hind[2 * t];
    int o = vind[2 * t + 1];
    float h0 = h[o * 64 + lane];
    float h1 = h[o * 64 + lane + 32];
#pragma unroll
    for (int r = 1; r < 16; ++r) {
        int m = s * r;
        float lw = lat2[t * 16 + r] / rowsum[m];
        unsafeAtomicAdd(&h2[(size_t)m * 64 + lane],      lw * h0);
        unsafeAtomicAdd(&h2[(size_t)m * 64 + lane + 32], lw * h1);
    }
    float lw0 = lat2[t * 16] / rowsum[0];
    atomicAdd(&acc0[lane],      lw0 * h0);
    atomicAdd(&acc0[lane + 32], lw0 * h1);
    __syncthreads();
    if (tid < 64) unsafeAtomicAdd(&h2[tid], acc0[tid]);
}

// ---------------------------------------------------------------------------
// K8: out (10000x16) = H2cat (10000x1024) @ weights2 (1024x16) + bias2
// via v_wmma_f32_16x16x4_f32, one wave per 16-row tile
// A frag: lane l: M=l%16, K=(l/16)*2+v ; B frag: n=l%16, K=(l/16)*2+v
// ka is always even and never crosses a 64-elem row -> single v2f A load
// ---------------------------------------------------------------------------
__global__ __launch_bounds__(256) void k_out(const float* __restrict__ h2,
                                             const float* __restrict__ w2,
                                             const float* __restrict__ bias2,
                                             float* __restrict__ out)
{
    int tile = blockIdx.x * 8 + (threadIdx.x >> 5);
    if (tile >= 625) return;                 // wave-uniform guard
    int lane = threadIdx.x & 31;
    int l16 = lane & 15, hi = lane >> 4;
    int n0 = tile * 16;
    const float* arow = h2 + (size_t)(n0 + l16) * 64;
    v8f acc = {};
    for (int k0 = 0; k0 < RP_ * E_; k0 += 4) {
        int ka = k0 + hi * 2;
        v2f a = *(const v2f*)&arow[(size_t)(ka >> 6) * (N_ * 64) + (ka & 63)];
        v2f b;
        b.x = w2[ka * 16 + l16];
        b.y = w2[ka * 16 + 16 + l16];
        acc = __builtin_amdgcn_wmma_f32_16x16x4_f32(false, a, false, b,
                                                    (short)0, acc, false, false);
    }
#pragma unroll
    for (int v = 0; v < 8; ++v) {
        int n = n0 + hi * 8 + v;
        out[n * 16 + l16] = acc[v] + bias2[l16];
    }
}

// ---------------------------------------------------------------------------
extern "C" void kernel_launch(void* const* d_in, const int* in_sizes, int n_in,
                              void* d_out, int out_size, void* d_ws, size_t ws_size,
                              hipStream_t stream)
{
    const float* nhots    = (const float*)d_in[0];
    const float* w1a      = (const float*)d_in[1];
    const float* b1a      = (const float*)d_in[2];
    const float* w1b      = (const float*)d_in[3];
    const float* b1b      = (const float*)d_in[4];
    const float* w2a      = (const float*)d_in[5];
    const float* b2a      = (const float*)d_in[6];
    const float* w2b      = (const float*)d_in[7];
    const float* b2b      = (const float*)d_in[8];
    const float* weights1 = (const float*)d_in[9];
    const float* bias1    = (const float*)d_in[10];
    const float* weights2 = (const float*)d_in[11];
    const float* bias2    = (const float*)d_in[12];
    const int*   hind     = (const int*)d_in[13];
    const int*   vind     = (const int*)d_in[14];
    float*       out      = (float*)d_out;

    char* ws = (char*)d_ws;                       // all offsets 256B-aligned
    _Float16* Bsw  = (_Float16*)(ws + 0);         //  2,048,000 B
    float* P1      = (float*)(ws +  2048000);     //  2,048,000 B
    float* P2      = (float*)(ws +  4096000);     //  2,048,000 B
    float* lat1    = (float*)(ws +  6144000);     //    512,000 B
    float* lat2    = (float*)(ws +  6656000);     //    512,000 B
    float* colsum  = (float*)(ws +  7168000);     //    640,000 B  } zeroed
    float* rowsum  = (float*)(ws +  7808000);     //    640,000 B  } region:
    float* h       = (float*)(ws +  8448000);     //  2,560,000 B  } 44.8 MB
    float* h2      = (float*)(ws + 11008000);     // 40,960,000 B  }
    (void)in_sizes; (void)n_in; (void)out_size; (void)ws_size;

    k_zero     <<<10938, 256, 0, stream>>>(colsum, 11200000L);
    k_buildB   <<< 4000, 256, 0, stream>>>(w1a, w2a, Bsw);
    k_stageA   <<<  250, 256, 0, stream>>>(nhots, Bsw, b1a, b2a, P1, P2);
    k_latents  <<<   32, 256, 0, stream>>>(P1, P2, w1b, b1b, w2b, b2b, lat1, lat2);
    k_normsum  <<<   32, 256, 0, stream>>>(lat1, lat2, hind, vind, colsum, rowsum);
    k_layer1   <<< 1000, 256, 0, stream>>>(lat1, colsum, hind, vind, weights1, h);
    k_relu_bias<<< 2500, 256, 0, stream>>>(h, bias1);
    k_layer2   <<< 1000, 256, 0, stream>>>(lat2, rowsum, hind, vind, h, h2);
    k_out      <<<   79, 256, 0, stream>>>(h2, weights2, bias2, out);
}
